// BasicNet_14499809592005
// MI455X (gfx1250) — matrix operand order repaired and checked by reading, not compiled, not run on hardware
//
#include <hip/hip_runtime.h>
#include <hip/hip_bf16.h>
#include <math.h>

// ---------------------------------------------------------------------------
// BasicNet (4x LEConv + mean-pool + MLP) for MI455X (gfx1250, wave32, WMMA)
//
// Layer l>=1 GEMM:  D[100000,384] = H_f16[100000,128] @ [W1|W2|W3]_f16 + bias
//   -> v_wmma_f32_16x16x32_f16 with TWO independent accumulator chains
//      (hides the 5-NOP WMMA->WMMA RAW hazard), weights pre-swizzled to
//      B-fragment layout and staged into LDS (96KB) with
//      global_load_async_to_lds_b128 (ASYNCcnt path).
// Edge phase: 1 wave per edge, lane = 4 channels, f32 atomics into agg.
// D (153.6MB) fits MI455X's 192MB L2 -> edge gathers are L2-resident.
//
// Workspace requirement ~231 MB.
// ---------------------------------------------------------------------------

#define N_NODES   100000
#define N_EDGES   600000
#define CH        128
#define NCAT      384            // 3*CH fused output columns
#define N_GRAPHS  1000

typedef __attribute__((ext_vector_type(16))) _Float16 v16h;
typedef __attribute__((ext_vector_type(8)))  float    v8f;
typedef __attribute__((ext_vector_type(4)))  _Float16 v4h;

// ---------------- prep: concatenated biases + layer-0 weight cat ------------
__global__ void prep_bias_kernel(const float* __restrict__ W1_0,
                                 const float* __restrict__ W2_0,
                                 const float* __restrict__ W3_0,
                                 const float* __restrict__ b1_0, const float* __restrict__ b3_0,
                                 const float* __restrict__ b1_1, const float* __restrict__ b3_1,
                                 const float* __restrict__ b1_2, const float* __restrict__ b3_2,
                                 const float* __restrict__ b1_3, const float* __restrict__ b3_3,
                                 float* __restrict__ Wcat0, float* __restrict__ bcat)
{
    int tid = blockIdx.x * blockDim.x + threadIdx.x;
    if (tid < 4 * NCAT) {                       // Wcat0: [4,384]
        int k = tid / NCAT, j = tid % NCAT;
        const float* W = (j < CH) ? W1_0 : (j < 2 * CH ? W2_0 : W3_0);
        Wcat0[tid] = W[k * CH + (j & (CH - 1))];
    } else if (tid < 4 * NCAT + 4 * NCAT) {     // bcat: 4 layers x 384
        int t = tid - 4 * NCAT;
        int l = t / NCAT, j = t % NCAT;
        const float* b1 = (l == 0) ? b1_0 : (l == 1) ? b1_1 : (l == 2) ? b1_2 : b1_3;
        const float* b3 = (l == 0) ? b3_0 : (l == 1) ? b3_1 : (l == 2) ? b3_2 : b3_3;
        bcat[t] = (j < CH) ? b1[j] : (j < 2 * CH ? 0.0f : b3[j - 2 * CH]);
    }
}

// ---------------- pack [W1|W2|W3] (f32, [128,128] each) into f16 B-fragments
// Fragment layout matches V_WMMA_F32_16X16X32_F16 B operand (32x16, wave32):
//   lanes 0-15 hold K = k0..k0+7,k0+16..k0+23 (N = lane), lanes 16-31 hold K = k0+8..k0+15,k0+24..k0+31.
// Linear index: (((ktile*24 + ntile)*32 + lane)*16 + h)
__global__ void pack_w_kernel(const float* __restrict__ W1,
                              const float* __restrict__ W2,
                              const float* __restrict__ W3,
                              _Float16* __restrict__ out)
{
    int tid = blockIdx.x * blockDim.x + threadIdx.x;
    if (tid >= 4 * 24 * 32 * 16) return;
    int h = tid & 15;       int t = tid >> 4;
    int lane = t & 31;      t >>= 5;
    int ntile = t % 24;     t /= 24;
    int ktile = t;                         // 0..3 (K chunks of 32)
    int n = ntile * 16 + (lane & 15);      // 0..383
    int k = ktile * 32 + ((lane >> 4) << 3) + (h & 7) + ((h >> 3) << 4);  // 0..127
    const float* W = (n < CH) ? W1 : (n < 2 * CH ? W2 : W3);
    out[tid] = (_Float16)W[k * CH + (n & (CH - 1))];
}

// ---------------- layer 0 (fan-in 4): D = x @ Wcat0 + bcat0 -----------------
__global__ void layer0_kernel(const float* __restrict__ x,
                              const float* __restrict__ Wcat0,
                              const float* __restrict__ bcat0,
                              float* __restrict__ D)
{
    size_t tid = (size_t)blockIdx.x * blockDim.x + threadIdx.x;
    if (tid >= (size_t)N_NODES * NCAT) return;
    int j = (int)(tid % NCAT);
    size_t n = tid / NCAT;
    const float4 xv = *(const float4*)(x + n * 4);
    float acc = bcat0[j];
    acc += xv.x * Wcat0[0 * NCAT + j];
    acc += xv.y * Wcat0[1 * NCAT + j];
    acc += xv.z * Wcat0[2 * NCAT + j];
    acc += xv.w * Wcat0[3 * NCAT + j];
    D[tid] = acc;
}

// ---------------- WMMA GEMM: D[M,384] = H_f16[M,128] @ Wpacked + bcat -------
// 256 threads = 8 waves; each wave owns a 16-row strip (block covers 128 rows).
// Wave: load 4 A-fragments once, loop 24 N-tiles, 4 WMMAs each (2 independent
// accumulator chains), store + bias.
__global__ void __launch_bounds__(256)
gemm_leconv_kernel(const _Float16* __restrict__ H,
                   const _Float16* __restrict__ Wp,   // packed fragments, this layer
                   const float* __restrict__ bcatL,   // 384
                   float* __restrict__ D, int M)
{
    extern __shared__ _Float16 smemW[];                // 128*384 halves = 96KB

    // Stage packed weights into LDS via CDNA5 async global->LDS DMA path.
    // LDS byte address = low 32 bits of the generic shared-aperture pointer.
    {
        const unsigned ldsBase = (unsigned)(uintptr_t)smemW;
        const char* gbase = (const char*)Wp;
        for (int i = threadIdx.x; i < (CH * NCAT * 2) / 16; i += blockDim.x) {
            unsigned loff = ldsBase + (unsigned)i * 16u;
            const char* gp = gbase + (size_t)i * 16u;
            asm volatile("global_load_async_to_lds_b128 %0, %1, off"
                         :: "v"(loff), "v"(gp) : "memory");
        }
        asm volatile("s_wait_asynccnt 0x0" ::: "memory");
    }
    __syncthreads();

    const int wave = threadIdx.x >> 5;
    const int lane = threadIdx.x & 31;
    const size_t m0 = (size_t)blockIdx.x * 128 + (size_t)wave * 16;
    if (m0 >= (size_t)M) return;

    union AF { uint4 u[2]; v16h h; };
    union BF { uint4 u[2]; v16h h; };

    // A fragments (16x32 f16, 4 K-chunks): lane<16 holds K=k0..k0+7,k0+16..k0+23
    // for row m0+lane; lane>=16 holds K=k0+8..k0+15,k0+24..k0+31 (ISA layout).
    v16h afrag[4];
    {
        const int row = (int)m0 + (lane & 15);
        const _Float16* hrow = H + (size_t)row * CH;
        #pragma unroll
        for (int kt = 0; kt < 4; ++kt) {
            const int kb = kt * 32 + ((lane & 16) ? 8 : 0);
            AF f;
            f.u[0] = *(const uint4*)(hrow + kb);        // 8 halves = 16B
            f.u[1] = *(const uint4*)(hrow + kb + 16);   // 8 halves = 16B
            afrag[kt] = f.h;
        }
    }

    const int rsel = (lane >> 4) << 3;   // D rows: lanes 16-31 hold M = r+8

    #pragma unroll 1
    for (int nt = 0; nt < 24; ++nt) {
        // Load all 4 B fragments for this N-tile from LDS.
        v16h bfrag[4];
        #pragma unroll
        for (int kt = 0; kt < 4; ++kt) {
            BF bf;
            const uint4* bp = (const uint4*)(smemW + ((size_t)(kt * 24 + nt) * 32 + lane) * 16);
            bf.u[0] = bp[0];
            bf.u[1] = bp[1];
            bfrag[kt] = bf.h;
        }
        // Two independent accumulator chains -> WMMA->WMMA hazard hidden by
        // alternating chains instead of v_nop stalls.
        v8f acc0 = {}, acc1 = {};
        acc0 = __builtin_amdgcn_wmma_f32_16x16x32_f16(false, afrag[0], false, bfrag[0],
                                                      (short)0, acc0, false, false);
        acc1 = __builtin_amdgcn_wmma_f32_16x16x32_f16(false, afrag[1], false, bfrag[1],
                                                      (short)0, acc1, false, false);
        acc0 = __builtin_amdgcn_wmma_f32_16x16x32_f16(false, afrag[2], false, bfrag[2],
                                                      (short)0, acc0, false, false);
        acc1 = __builtin_amdgcn_wmma_f32_16x16x32_f16(false, afrag[3], false, bfrag[3],
                                                      (short)0, acc1, false, false);

        const int n = nt * 16 + (lane & 15);
        const float bv = bcatL[n];
        float* dcol = D + (m0 + (size_t)rsel) * NCAT + n;
        #pragma unroll
        for (int r = 0; r < 8; ++r)
            dcol[(size_t)r * NCAT] = acc0[r] + acc1[r] + bv;
    }
}

// ---------------- generic float fill ---------------------------------------
__global__ void fill_kernel(float* __restrict__ p, float v, size_t n)
{
    size_t tid = (size_t)blockIdx.x * blockDim.x + threadIdx.x;
    if (tid < n) p[tid] = v;
}

// ---------------- edge scatter: agg[dst] += (A[src]-B[dst]) * sigmoid(s) ----
// One wave per edge: uniform src/dst, lane owns 4 channels (float4).
__global__ void edge_kernel(const float* __restrict__ D,
                            const long long* __restrict__ ei,
                            const float* __restrict__ escore,
                            int layer,
                            float* __restrict__ agg)
{
    const int lane = threadIdx.x & 31;
    const size_t e = (size_t)blockIdx.x * 8 + (threadIdx.x >> 5);
    if (e >= N_EDGES) return;
    const int src = (int)ei[e];
    const int dst = (int)ei[(size_t)N_EDGES + e];
    const float s = escore[e * 4 + layer];
    const float ew = 1.0f / (1.0f + __expf(-s));
    const float4 a = ((const float4*)(D + (size_t)src * NCAT))[lane];        // A part
    const float4 b = ((const float4*)(D + (size_t)dst * NCAT + CH))[lane];   // B part
    float* o = agg + (size_t)dst * CH + lane * 4;
    atomicAdd(o + 0, (a.x - b.x) * ew);
    atomicAdd(o + 1, (a.y - b.y) * ew);
    atomicAdd(o + 2, (a.z - b.z) * ew);
    atomicAdd(o + 3, (a.w - b.w) * ew);
}

// ---------------- h_next = relu(agg + C3), repack as f16 for next GEMM ------
__global__ void relu_next_kernel(const float* __restrict__ agg,
                                 const float* __restrict__ D,
                                 _Float16* __restrict__ Hf16)
{
    size_t idx = (size_t)blockIdx.x * blockDim.x + threadIdx.x;   // node*32 + c4
    if (idx >= (size_t)N_NODES * 32) return;
    const size_t node = idx >> 5;
    const int c4 = (int)(idx & 31);
    const float4 g = ((const float4*)agg)[idx];
    const float* c3 = D + node * NCAT + 2 * CH + c4 * 4;
    v4h o;
    o[0] = (_Float16)fmaxf(g.x + c3[0], 0.0f);
    o[1] = (_Float16)fmaxf(g.y + c3[1], 0.0f);
    o[2] = (_Float16)fmaxf(g.z + c3[2], 0.0f);
    o[3] = (_Float16)fmaxf(g.w + c3[3], 0.0f);
    *(v4h*)(Hf16 + idx * 4) = o;
}

// ---------------- last layer: relu fused with graph sum pooling -------------
__global__ void relu_pool_kernel(const float* __restrict__ agg,
                                 const float* __restrict__ D,
                                 const long long* __restrict__ batch,
                                 float* __restrict__ gf_sums,   // [1000,128]
                                 float* __restrict__ counts)    // [1000]
{
    size_t idx = (size_t)blockIdx.x * blockDim.x + threadIdx.x;
    if (idx >= (size_t)N_NODES * 32) return;
    const size_t node = idx >> 5;
    const int c4 = (int)(idx & 31);
    const float4 g = ((const float4*)agg)[idx];
    const float* c3 = D + node * NCAT + 2 * CH + c4 * 4;
    const int grp = (int)batch[node];
    float* o = gf_sums + (size_t)grp * CH + c4 * 4;
    atomicAdd(o + 0, fmaxf(g.x + c3[0], 0.0f));
    atomicAdd(o + 1, fmaxf(g.y + c3[1], 0.0f));
    atomicAdd(o + 2, fmaxf(g.z + c3[2], 0.0f));
    atomicAdd(o + 3, fmaxf(g.w + c3[3], 0.0f));
    if (c4 == 0) atomicAdd(counts + grp, 1.0f);
}

// ---------------- mean divide (in place on d_out graph_feature) -------------
__global__ void divide_kernel(float* __restrict__ gf, const float* __restrict__ counts)
{
    int tid = blockIdx.x * blockDim.x + threadIdx.x;
    if (tid >= N_GRAPHS * CH) return;
    gf[tid] = gf[tid] / fmaxf(counts[tid >> 7], 1.0f);
}

// ---------------- final MLP: pred = relu(gf@Wm1+bm1)@Wm2+bm2 ----------------
__global__ void __launch_bounds__(256)
mlp_kernel(const float* __restrict__ gf,
           const float* __restrict__ Wm1, const float* __restrict__ bm1,
           const float* __restrict__ Wm2, const float* __restrict__ bm2,
           float* __restrict__ pred)
{
    __shared__ float gfs[CH];
    __shared__ float hid[2 * CH];
    const int g = blockIdx.x;
    const int t = threadIdx.x;
    if (t < CH) gfs[t] = gf[(size_t)g * CH + t];
    __syncthreads();
    float acc = bm1[t];
    #pragma unroll 4
    for (int k = 0; k < CH; ++k) acc += gfs[k] * Wm1[k * (2 * CH) + t];
    hid[t] = fmaxf(acc, 0.0f);
    __syncthreads();
    if (t < 2) {
        float p = bm2[t];
        for (int j = 0; j < 2 * CH; ++j) p += hid[j] * Wm2[j * 2 + t];
        pred[(size_t)g * 2 + t] = p;
    }
}

// ---------------------------------------------------------------------------
extern "C" void kernel_launch(void* const* d_in, const int* in_sizes, int n_in,
                              void* d_out, int out_size, void* d_ws, size_t ws_size,
                              hipStream_t stream)
{
    (void)in_sizes; (void)n_in; (void)out_size; (void)ws_size;

    const float*     x      = (const float*)d_in[0];
    const long long* ei     = (const long long*)d_in[1];
    const long long* batch  = (const long long*)d_in[2];
    const float*     escore = (const float*)d_in[3];
    // d_in[4] = num_graphs scalar (constant 1000, unused)
    const float* cw[4][5];   // per layer: W1, b1, W2, W3, b3
    for (int l = 0; l < 4; ++l)
        for (int p = 0; p < 5; ++p)
            cw[l][p] = (const float*)d_in[5 + 5 * l + p];
    const float* Wm1 = (const float*)d_in[25];
    const float* bm1 = (const float*)d_in[26];
    const float* Wm2 = (const float*)d_in[27];
    const float* bm2 = (const float*)d_in[28];

    // ---- workspace carve-up (all offsets 16B-aligned) ----
    char* w = (char*)d_ws;
    float* D = (float*)w;            w += (size_t)N_NODES * NCAT * sizeof(float);   // 153.6 MB
    float* agg = (float*)w;          w += (size_t)N_NODES * CH * sizeof(float);     // 51.2 MB
    _Float16* Hf16 = (_Float16*)w;   w += (size_t)N_NODES * CH * sizeof(_Float16);  // 25.6 MB
    _Float16* Wp = (_Float16*)w;     w += (size_t)3 * CH * NCAT * sizeof(_Float16); // 288 KB
    float* Wcat0 = (float*)w;        w += (size_t)4 * NCAT * sizeof(float);
    float* bcat = (float*)w;         w += (size_t)4 * NCAT * sizeof(float);
    float* counts = (float*)w;       w += (size_t)N_GRAPHS * sizeof(float);

    float* gf   = (float*)d_out;                     // [1000,128]
    float* pred = (float*)d_out + N_GRAPHS * CH;     // [1000,2]

    // ---- prep: biases + layer-0 weight cat + f16 B-fragment packing ----
    prep_bias_kernel<<<(4 * NCAT * 2 + 255) / 256, 256, 0, stream>>>(
        cw[0][0], cw[0][2], cw[0][3],
        cw[0][1], cw[0][4], cw[1][1], cw[1][4],
        cw[2][1], cw[2][4], cw[3][1], cw[3][4],
        Wcat0, bcat);
    for (int l = 1; l <= 3; ++l)
        pack_w_kernel<<<(4 * 24 * 32 * 16) / 256, 256, 0, stream>>>(
            cw[l][0], cw[l][2], cw[l][3], Wp + (size_t)(l - 1) * CH * NCAT);

    // ---- layer 0 fused projections (fan-in 4) ----
    layer0_kernel<<<(int)(((size_t)N_NODES * NCAT + 255) / 256), 256, 0, stream>>>(
        x, Wcat0, bcat, D);

    // ---- 4 LEConv layers ----
    for (int l = 0; l < 4; ++l) {
        if (l > 0) {
            gemm_leconv_kernel<<<(N_NODES + 127) / 128, 256,
                                 CH * NCAT * sizeof(_Float16), stream>>>(
                Hf16, Wp + (size_t)(l - 1) * CH * NCAT, bcat + l * NCAT, D, N_NODES);
        }
        fill_kernel<<<(int)(((size_t)N_NODES * CH + 255) / 256), 256, 0, stream>>>(
            agg, 0.0f, (size_t)N_NODES * CH);
        edge_kernel<<<N_EDGES / 8, 256, 0, stream>>>(D, ei, escore, l, agg);
        if (l < 3) {
            relu_next_kernel<<<(int)(((size_t)N_NODES * 32 + 255) / 256), 256, 0, stream>>>(
                agg, D, Hf16);
        } else {
            fill_kernel<<<(N_GRAPHS * CH + 255) / 256, 256, 0, stream>>>(
                gf, 0.0f, (size_t)N_GRAPHS * CH);
            fill_kernel<<<(N_GRAPHS + 255) / 256, 256, 0, stream>>>(
                counts, 0.0f, (size_t)N_GRAPHS);
            relu_pool_kernel<<<(int)(((size_t)N_NODES * 32 + 255) / 256), 256, 0, stream>>>(
                agg, D, batch, gf, counts);
        }
    }

    // ---- mean pooling + MLP head ----
    divide_kernel<<<(N_GRAPHS * CH + 255) / 256, 256, 0, stream>>>(gf, counts);
    mlp_kernel<<<N_GRAPHS, 256, 0, stream>>>(gf, Wm1, bm1, Wm2, bm2, pred);
}
